// Attention_78838419685807
// MI455X (gfx1250) — compile-verified, matrix operand order checked by reading, not run on hardware
//
#include <hip/hip_runtime.h>
#include <hip/hip_bf16.h>

// ---------------------------------------------------------------------------
// MI455X (gfx1250, wave32) implementation.
// Heavy math: v_wmma_f32_16x16x32_bf16 (f32 accumulate).
// Tile staging: global_load_async_to_lds_b128 (ASYNCcnt) wherever no
// transpose is needed; manual transpose path only for the GEMM B-tile.
// Key identity: centered-dot = dot - d*mu_q*mu_k, so the covariance term
// reuses the Q.K^T GEMM -> one NxN GEMM per (h,b) plus rank-1 corrections.
// ---------------------------------------------------------------------------

typedef __attribute__((ext_vector_type(16))) __bf16        v16bf;
typedef __attribute__((ext_vector_type(8)))  float         v8f;
typedef __attribute__((ext_vector_type(8)))  unsigned int  v8u;

#define TOKENS   16384          // B*N = 32*512
#define DIMC     512
#define NHEADS   8
#define DHEAD    64
#define HS       (NHEADS * TOKENS)   // 131072 entries per stat array

// ---- helpers ---------------------------------------------------------------

static __device__ __forceinline__ unsigned short f2bf(float f) {
  unsigned u = __builtin_bit_cast(unsigned, f);
  unsigned r = u + 0x7FFFu + ((u >> 16) & 1u);   // round-to-nearest-even
  return (unsigned short)(r >> 16);
}

static __device__ __forceinline__ float bf2f(unsigned short h) {
  return __builtin_bit_cast(float, (unsigned)h << 16);
}

static __device__ __forceinline__ v8f zero_v8f() {
  v8f z;
#pragma unroll
  for (int i = 0; i < 8; ++i) z[i] = 0.f;
  return z;
}

// Async global->LDS 16-byte copy (CDNA5, ASYNCcnt-tracked, GVS addressing:
// uniform SGPR64 base + per-lane VGPR byte offset; VDST VGPR = LDS offset).
static __device__ __forceinline__ void async_cp_b128(unsigned short* lds_dst,
                                                     const unsigned short* gbase,
                                                     unsigned byte_off) {
  unsigned lds_off = (unsigned)(size_t)lds_dst;  // flat LDS addr low 32b = offset
  asm volatile("global_load_async_to_lds_b128 %0, %1, %2"
               :: "v"(lds_off), "v"(byte_off), "s"(gbase)
               : "memory");
}

static __device__ __forceinline__ void wait_async() {
  asm volatile("s_wait_asynccnt 0x0" ::: "memory");
}

// A-matrix 16x32 bf16 fragment (ISA 7.12.2): lane L holds row L%16,
// K-halves kb = 8*(L/16): VGPR p<4 -> K = kb+2p, kb+2p+1 ; p>=4 -> K = kb+16+...
// Source LDS layout: [rows][K] (pairs along K contiguous -> b32/b128 LDS reads).
// Same mapping serves the B operand when LDS holds B transposed as [N][K].
static __device__ __forceinline__ v16bf ld_frag_kmajor(const unsigned short* lds,
                                                       int row, int stride, int k0) {
  int lane = threadIdx.x & 31;
  const unsigned short* p = lds + (size_t)(row + (lane & 15)) * stride
                                + k0 + ((lane >> 4) << 3);
  v8u u;
#pragma unroll
  for (int i = 0; i < 8; ++i) {
    int k = (i < 4) ? (2 * i) : (16 + 2 * (i - 4));
    u[i] = *(const unsigned int*)(p + k);
  }
  return __builtin_bit_cast(v16bf, u);
}

// B operand from LDS laid out naturally as [K][N] (stride strideN): gather
// the per-lane K pattern with 16-bit LDS reads.
static __device__ __forceinline__ v16bf ld_frag_nmajor(const unsigned short* lds,
                                                       int k0, int col, int strideN) {
  int lane = threadIdx.x & 31;
  int c  = col + (lane & 15);
  int kb = k0 + ((lane >> 4) << 3);
  v8u u;
#pragma unroll
  for (int i = 0; i < 8; ++i) {
    int k = kb + ((i < 4) ? (2 * i) : (16 + 2 * (i - 4)));
    unsigned lo = lds[(size_t)k * strideN + c];
    unsigned hi = lds[(size_t)(k + 1) * strideN + c];
    u[i] = lo | (hi << 16);
  }
  return __builtin_bit_cast(v16bf, u);
}

static __device__ __forceinline__ v8f wmma_bf16(v16bf a, v16bf b, v8f c) {
  return __builtin_amdgcn_wmma_f32_16x16x32_bf16(false, a, false, b,
                                                 (short)0, c, false, false);
}

// ---- kernel 0: weights f32 -> bf16 ----------------------------------------

__global__ void cvt_w_kernel(const float* __restrict__ win,
                             const float* __restrict__ wout,
                             unsigned short* __restrict__ obin,
                             unsigned short* __restrict__ obout) {
  int idx = blockIdx.x * 256 + threadIdx.x;
  const int n = DIMC * DIMC;
  if (idx < n) obin[idx] = f2bf(win[idx]);
  else         obout[idx - n] = f2bf(wout[idx - n]);
}

// ---- kernel 1: LayerNorm -> bf16 (one wave32 per 512-wide row) -------------

__global__ __launch_bounds__(256)
void ln_bf16_kernel(const float* __restrict__ x, const float* __restrict__ w,
                    const float* __restrict__ b, unsigned short* __restrict__ out) {
  int wave = threadIdx.x >> 5, lane = threadIdx.x & 31;
  int row = blockIdx.x * 8 + wave;
  const float* xr = x + (size_t)row * DIMC + lane * 16;
  float v[16];
  float s = 0.f, ss = 0.f;
#pragma unroll
  for (int i = 0; i < 4; ++i) {
    float4 t = ((const float4*)xr)[i];
    v[4*i+0] = t.x; v[4*i+1] = t.y; v[4*i+2] = t.z; v[4*i+3] = t.w;
  }
#pragma unroll
  for (int i = 0; i < 16; ++i) { s += v[i]; ss += v[i] * v[i]; }
#pragma unroll
  for (int off = 16; off >= 1; off >>= 1) {      // wave32 reduction
    s  += __shfl_xor(s,  off);
    ss += __shfl_xor(ss, off);
  }
  float mean = s * (1.f / 512.f);
  float var  = ss * (1.f / 512.f) - mean * mean; // biased, as nn.LayerNorm
  float rstd = rsqrtf(var + 1e-5f);
  int c0 = lane * 16;
  unsigned short h[16];
#pragma unroll
  for (int i = 0; i < 16; ++i)
    h[i] = f2bf((v[i] - mean) * rstd * w[c0 + i] + b[c0 + i]);
  uint4* dst = (uint4*)(out + (size_t)row * DIMC + c0);
  dst[0] = *(uint4*)(h);
  dst[1] = *(uint4*)(h + 8);
}

// ---- kernel 2/5: generic 512-wide bf16 GEMM, 128x128 tile, BK=64 ----------
// 8 waves: wave = (wm 0..3, wn 0..1) -> 32x64 sub-tile = 2x4 WMMA accums.
// A tile staged with async global->LDS; B tile transposed through VGPRs.

template <bool OUT_BF16>
__global__ __launch_bounds__(256)
void gemm512_kernel(const unsigned short* __restrict__ A,
                    const unsigned short* __restrict__ Bm,
                    const float* __restrict__ bias,
                    void* __restrict__ out) {
  __shared__ alignas(16) unsigned short As[128 * 64];   // [row][k]
  __shared__ alignas(16) unsigned short Bs[128 * 64];   // [col][k]  (transposed)
  int t = threadIdx.x;
  int wave = t >> 5;
  int wm = wave >> 1, wn = wave & 1;
  int rowBase = blockIdx.y * 128, colBase = blockIdx.x * 128;

  v8f acc[2][4];
#pragma unroll
  for (int s = 0; s < 2; ++s)
#pragma unroll
    for (int j = 0; j < 4; ++j) acc[s][j] = zero_v8f();

  for (int k0 = 0; k0 < 512; k0 += 64) {
    { // A tile: 128x64 halves, 4 async b128 per thread
      int r = t >> 1, cs = (t & 1) * 32;
      unsigned goff = (unsigned)(((size_t)(rowBase + r) * 512 + k0 + cs) * 2);
      unsigned short* ldst = As + r * 64 + cs;
#pragma unroll
      for (int i = 0; i < 4; ++i)
        async_cp_b128(ldst + i * 8, A, goff + i * 16);
    }
    { // B tile: read [k][n] rows coalesced, store transposed [n][k]
      int kk = t >> 2, ns = (t & 3) * 32;
      const uint4* src = (const uint4*)(Bm + (size_t)(k0 + kk) * 512 + colBase + ns);
      uint4 b0 = src[0], b1 = src[1], b2 = src[2], b3 = src[3];
      unsigned short hb[32];
      *(uint4*)(hb)      = b0; *(uint4*)(hb + 8)  = b1;
      *(uint4*)(hb + 16) = b2; *(uint4*)(hb + 24) = b3;
#pragma unroll
      for (int i = 0; i < 32; ++i) Bs[(ns + i) * 64 + kk] = hb[i];
    }
    wait_async();
    __syncthreads();
#pragma unroll
    for (int kk = 0; kk < 64; kk += 32) {
      v16bf afr[2], bfr[4];
#pragma unroll
      for (int s = 0; s < 2; ++s) afr[s] = ld_frag_kmajor(As, wm * 32 + s * 16, 64, kk);
#pragma unroll
      for (int j = 0; j < 4; ++j) bfr[j] = ld_frag_kmajor(Bs, wn * 64 + j * 16, 64, kk);
#pragma unroll
      for (int s = 0; s < 2; ++s)
#pragma unroll
        for (int j = 0; j < 4; ++j)
          acc[s][j] = wmma_bf16(afr[s], bfr[j], acc[s][j]);
    }
    __syncthreads();
  }

  int lane = t & 31;
  int rOff = (lane >> 4) * 8, cOff = lane & 15;   // C/D layout (ISA 7.12.2)
#pragma unroll
  for (int s = 0; s < 2; ++s)
#pragma unroll
    for (int j = 0; j < 4; ++j) {
      int col = colBase + wn * 64 + j * 16 + cOff;
#pragma unroll
      for (int r = 0; r < 8; ++r) {
        int row = rowBase + wm * 32 + s * 16 + rOff + r;
        float vv = acc[s][j][r];
        if (OUT_BF16)
          ((unsigned short*)out)[(size_t)row * 512 + col] = f2bf(vv);
        else
          ((float*)out)[(size_t)row * 512 + col] = vv + bias[col];
      }
    }
}

// ---- kernel 3: per-(head,token) moments of f_q / f_k -----------------------
// One wave per token; 4 lanes cover each head's 64 values.

__global__ __launch_bounds__(256)
void stats_kernel(const unsigned short* __restrict__ Fq,
                  const unsigned short* __restrict__ Fk,
                  float* __restrict__ sq, float* __restrict__ sk) {
  int wave = threadIdx.x >> 5, lane = threadIdx.x & 31;
  int id = blockIdx.x * 8 + wave;        // 0..32767
  int tensor = id >> 14;                 // 0 = q, 1 = k
  int token  = id & (TOKENS - 1);
  const unsigned short* F = tensor ? Fk : Fq;
  const unsigned short* row = F + (size_t)token * DIMC + lane * 16;
  float s = 0.f, ss = 0.f;
#pragma unroll
  for (int i = 0; i < 16; ++i) {
    float v = bf2f(row[i]);
    s += v; ss += v * v;
  }
  s += __shfl_xor(s, 1); ss += __shfl_xor(ss, 1);
  s += __shfl_xor(s, 2); ss += __shfl_xor(ss, 2);
  if ((lane & 3) == 0) {
    int h = lane >> 2;
    float mu  = s * (1.f / 64.f);
    float inv = rsqrtf(ss);                           // 1/||f||
    float var = (ss - 64.f * mu * mu) * (1.f / 63.f); // unbiased
    float* dst = tensor ? sk : sq;
    size_t idx = (size_t)h * TOKENS + token;
    dst[idx]                  = inv;
    dst[(size_t)HS + idx]     = mu;
    dst[2 * (size_t)HS + idx] = var;
  }
}

// ---- kernel 4: attention (no softmax), per (h, b, 128-row n-block) ---------
// Phase 1: S = Q K^T (WMMA).  Elementwise score transform in f32 -> bf16 LDS.
// Phase 2: O += S V (WMMA, persistent f32 accumulators).  m streamed by 64.
// Q/K/V tiles staged with async global->LDS (no transpose needed: natural
// [token][d] layout serves both k-major and n-major fragment loaders).

__global__ __launch_bounds__(256)
void attn_kernel(const unsigned short* __restrict__ Fq,
                 const unsigned short* __restrict__ Fk,
                 const unsigned short* __restrict__ Fv,
                 const float* __restrict__ sq, const float* __restrict__ sk,
                 const float* __restrict__ cov_raw, const float* __restrict__ var_raw,
                 unsigned short* __restrict__ attn_out) {
  __shared__ alignas(16) unsigned short Qs[128 * 64];  // [n][d]
  __shared__ alignas(16) unsigned short KVs[64 * 64];  // [m][d] (K, then V)
  __shared__ alignas(16) unsigned short Ss[128 * 64];  // [n][m] bf16 scores
  __shared__ float rowInv[128], rowMu[128], rowVar[128];
  __shared__ float colInv[64],  colMu[64],  colVar[64];

  int t = threadIdx.x;
  int wave = t >> 5, lane = t & 31;
  int nb = blockIdx.x, bb = blockIdx.y, hh = blockIdx.z;
  int tb = bb * 512;
  int n0 = nb * 128;
  int hcol = hh * DHEAD;

  float cov_w = 1.f / (1.f + __expf(-cov_raw[0]));
  float var_w = 1.f / (1.f + __expf(-var_raw[0]));
  float cos_w = 1.f - cov_w - var_w;
  float cov_d = cov_w * (1.f / 64.f);
  float var_d = var_w * (1.f / 64.f);

  { // Q tile: 128x64 halves, 4 async b128 per thread
    int r = t >> 1, ds = (t & 1) * 32;
    unsigned goff = (unsigned)(((size_t)(tb + n0 + r) * 512 + hcol + ds) * 2);
    unsigned short* ldst = Qs + r * 64 + ds;
#pragma unroll
    for (int i = 0; i < 4; ++i)
      async_cp_b128(ldst + i * 8, Fq, goff + i * 16);
  }
  if (t < 128) {
    size_t idx = (size_t)hh * TOKENS + tb + n0 + t;
    rowInv[t] = sq[idx];
    rowMu[t]  = sq[(size_t)HS + idx];
    rowVar[t] = sq[2 * (size_t)HS + idx];
  }
  wait_async();
  __syncthreads();

  int wm   = wave >> 1;   // 0..3 : 32-row n slice
  int wsub = wave & 1;    // phase1: 32-col m half; phase2: 32-col d half
  int rOff = (lane >> 4) * 8, cOff = lane & 15;

  v8f acc_o[2][2];
#pragma unroll
  for (int s = 0; s < 2; ++s)
#pragma unroll
    for (int j = 0; j < 2; ++j) acc_o[s][j] = zero_v8f();

  for (int mb = 0; mb < 8; ++mb) {
    int m0 = mb * 64;
    { // K tile 64x64: 2 async b128 per thread
      int r = t >> 2, ds = (t & 3) * 16;
      unsigned goff = (unsigned)(((size_t)(tb + m0 + r) * 512 + hcol + ds) * 2);
      unsigned short* ldst = KVs + r * 64 + ds;
      async_cp_b128(ldst,     Fk, goff);
      async_cp_b128(ldst + 8, Fk, goff + 16);
    }
    if (t < 64) {
      size_t idx = (size_t)hh * TOKENS + tb + m0 + t;
      colInv[t] = sk[idx];
      colMu[t]  = sk[(size_t)HS + idx];
      colVar[t] = sk[2 * (size_t)HS + idx];
    }
    wait_async();
    __syncthreads();

    // Phase 1: S = Q K^T   (wave covers 32n x 32m; K over d = 2 steps)
    v8f sacc[2][2];
#pragma unroll
    for (int s = 0; s < 2; ++s)
#pragma unroll
      for (int j = 0; j < 2; ++j) sacc[s][j] = zero_v8f();
#pragma unroll
    for (int kk = 0; kk < 64; kk += 32) {
      v16bf aq[2], bk[2];
#pragma unroll
      for (int s = 0; s < 2; ++s) aq[s] = ld_frag_kmajor(Qs, wm * 32 + s * 16, 64, kk);
#pragma unroll
      for (int j = 0; j < 2; ++j) bk[j] = ld_frag_kmajor(KVs, wsub * 32 + j * 16, 64, kk);
#pragma unroll
      for (int s = 0; s < 2; ++s)
#pragma unroll
        for (int j = 0; j < 2; ++j)
          sacc[s][j] = wmma_bf16(aq[s], bk[j], sacc[s][j]);
    }

    // Elementwise score transform (cos/cov/var blend) -> bf16 in LDS.
#pragma unroll
    for (int s = 0; s < 2; ++s)
#pragma unroll
      for (int j = 0; j < 2; ++j) {
        int ml = wsub * 32 + j * 16 + cOff;
        float ci = colInv[ml], cm = colMu[ml], cv = colVar[ml];
#pragma unroll
        for (int r = 0; r < 8; ++r) {
          int nl = wm * 32 + s * 16 + rOff + r;
          float dv = sacc[s][j][r];
          float sv = dv * (cos_w * rowInv[nl] * ci + cov_d)
                   + var_d * rowVar[nl] * cv
                   - cov_w * rowMu[nl]  * cm;
          Ss[nl * 64 + ml] = f2bf(sv);
        }
      }
    __syncthreads();

    { // V tile 64x64 into KVs (phase-1 K reads completed at barrier above)
      int r = t >> 2, ds = (t & 3) * 16;
      unsigned goff = (unsigned)(((size_t)(tb + m0 + r) * 512 + hcol + ds) * 2);
      unsigned short* ldst = KVs + r * 64 + ds;
      async_cp_b128(ldst,     Fv, goff);
      async_cp_b128(ldst + 8, Fv, goff + 16);
    }
    wait_async();
    __syncthreads();

    // Phase 2: O += S V   (wave covers 32n x 32d; K over m = 2 steps)
#pragma unroll
    for (int kk = 0; kk < 64; kk += 32) {
      v16bf asf[2], bvf[2];
#pragma unroll
      for (int s = 0; s < 2; ++s) asf[s] = ld_frag_kmajor(Ss, wm * 32 + s * 16, 64, kk);
#pragma unroll
      for (int j = 0; j < 2; ++j) bvf[j] = ld_frag_nmajor(KVs, kk, wsub * 32 + j * 16, 64);
#pragma unroll
      for (int s = 0; s < 2; ++s)
#pragma unroll
        for (int j = 0; j < 2; ++j)
          acc_o[s][j] = wmma_bf16(asf[s], bvf[j], acc_o[s][j]);
    }
    __syncthreads();
  }

  // Epilogue: write 'b n (h d)' layout in bf16 for the output projection.
#pragma unroll
  for (int s = 0; s < 2; ++s)
#pragma unroll
    for (int j = 0; j < 2; ++j) {
      int col = hcol + wsub * 32 + j * 16 + cOff;
#pragma unroll
      for (int r = 0; r < 8; ++r) {
        int row = tb + n0 + wm * 32 + s * 16 + rOff + r;
        attn_out[(size_t)row * 512 + col] = f2bf(acc_o[s][j][r]);
      }
    }
}

// ---------------------------------------------------------------------------

extern "C" void kernel_launch(void* const* d_in, const int* in_sizes, int n_in,
                              void* d_out, int out_size, void* d_ws, size_t ws_size,
                              hipStream_t stream) {
  (void)in_sizes; (void)n_in; (void)out_size; (void)ws_size;

  const float* q       = (const float*)d_in[0];
  const float* k       = (const float*)d_in[1];
  const float* v       = (const float*)d_in[2];
  const float* ln_w    = (const float*)d_in[3];
  const float* ln_b    = (const float*)d_in[4];
  const float* W_in    = (const float*)d_in[5];
  const float* W_out   = (const float*)d_in[6];
  const float* b_out   = (const float*)d_in[7];
  const float* cov_raw = (const float*)d_in[8];
  const float* var_raw = (const float*)d_in[9];

  // Workspace layout (~85 MB): bf16 weights, LN scratch, bf16 features,
  // per-token stats, bf16 attention output.
  char* ws = (char*)d_ws;
  size_t o = 0;
  auto alloc = [&](size_t bytes) -> char* {
    char* p = ws + o;
    o = (o + bytes + 255) & ~(size_t)255;
    return p;
  };
  unsigned short* Win_bf  = (unsigned short*)alloc((size_t)DIMC * DIMC * 2);
  unsigned short* Wout_bf = (unsigned short*)alloc((size_t)DIMC * DIMC * 2);
  unsigned short* xn      = (unsigned short*)alloc((size_t)TOKENS * DIMC * 2);
  unsigned short* Fq      = (unsigned short*)alloc((size_t)TOKENS * DIMC * 2);
  unsigned short* Fk      = (unsigned short*)alloc((size_t)TOKENS * DIMC * 2);
  unsigned short* Fv      = (unsigned short*)alloc((size_t)TOKENS * DIMC * 2);
  float*          sq      = (float*)alloc((size_t)3 * HS * 4);
  float*          sk      = (float*)alloc((size_t)3 * HS * 4);
  unsigned short* attn    = (unsigned short*)alloc((size_t)TOKENS * DIMC * 2);

  cvt_w_kernel<<<2048, 256, 0, stream>>>(W_in, W_out, Win_bf, Wout_bf);

  const float* xs[3]    = {q, k, v};
  unsigned short* Fs[3] = {Fq, Fk, Fv};
  for (int i = 0; i < 3; ++i) {
    ln_bf16_kernel<<<2048, 256, 0, stream>>>(xs[i], ln_w, ln_b, xn);
    gemm512_kernel<true><<<dim3(4, 128), 256, 0, stream>>>(xn, Win_bf, nullptr, Fs[i]);
  }

  stats_kernel<<<4096, 256, 0, stream>>>(Fq, Fk, sq, sk);

  attn_kernel<<<dim3(4, 32, 8), 256, 0, stream>>>(Fq, Fk, Fv, sq, sk,
                                                  cov_raw, var_raw, attn);

  gemm512_kernel<false><<<dim3(4, 128), 256, 0, stream>>>(attn, Wout_bf, b_out, d_out);
}
